// RNNNet_20349555048770
// MI455X (gfx1250) — matrix operand order repaired and checked by reading, not compiled, not run on hardware
//
#include <hip/hip_runtime.h>
#include <hip/hip_bf16.h>

#define INPUT_ 28
#define HID_   128
#define OUT_   10
#define BATCH_ 65536
#define T_     28
#define KCAT   160           // 128 (h) + 32 (x padded from 28): K of fused GEMM
#define RPW    32            // rows per wave (two 16-row m-tiles)
#define WAVES_ 4
#define RPB    (RPW * WAVES_)   // 128 rows per block
#define TWOLOG2E 2.88539008177792681472f

typedef __attribute__((ext_vector_type(16))) _Float16 v16h;
typedef __attribute__((ext_vector_type(8)))  _Float16 v8h;
typedef __attribute__((ext_vector_type(8)))  float    v8f;
typedef __attribute__((ext_vector_type(4)))  float    v4f;

__device__ inline v16h combine16(v8h lo, v8h hi) {
    v16h a;
#pragma unroll
    for (int i = 0; i < 8; ++i) { a[i] = lo[i]; a[i + 8] = hi[i]; }
    return a;
}

__global__ __launch_bounds__(128) void rnn_fused_wmma(
    const float* __restrict__ x,    const float* __restrict__ w_ih,
    const float* __restrict__ w_hh, const float* __restrict__ b_ih,
    const float* __restrict__ b_hh, const float* __restrict__ fc_w,
    const float* __restrict__ fc_b, float* __restrict__ out)
{
    // Fused A-matrix (weights): sW[n*KCAT + k] = k<128 ? w_hh[n][k] : w_ih[n][k-128], 0-pad
    __shared__ _Float16 sW[HID_ * KCAT];              // 40 KB
    __shared__ float    sBiasK[HID_];                 // (b_ih+b_hh)*2log2e
    __shared__ _Float16 sH[WAVES_ * RPW * HID_];      // 32 KB, per-wave h tiles [32][128]

    const int tid    = threadIdx.x;
    const int wave   = tid >> 5;
    const int lane   = tid & 31;
    const int l15    = lane & 15;
    const int hiHalf = lane >> 4;      // 0: lanes 0-15, 1: lanes 16-31

    // ---- stage fused weights (f32 -> f16), vectorized: 2560 v8h chunks ----
    for (int i = tid; i < (HID_ * KCAT) / 8; i += 128) {
        int n  = i / (KCAT / 8);
        int cr = i - n * (KCAT / 8);               // chunk-in-row 0..19
        v4f q0, q1;
        if (cr < 16) {                              // w_hh part: k = cr*8 .. +7
            const v4f* src = (const v4f*)(w_hh + n * HID_ + cr * 8);
            q0 = src[0]; q1 = src[1];
        } else {                                    // w_ih part: k-128 = (cr-16)*8
            int base = (cr - 16) * 8;
            const v4f* src = (const v4f*)(w_ih + n * INPUT_ + base);
            q0 = src[0];
            q1 = (cr == 19) ? (v4f){0.f, 0.f, 0.f, 0.f} : src[1];  // cols 28..31 = 0
        }
        v8h e;
#pragma unroll
        for (int j = 0; j < 4; ++j) { e[j] = (_Float16)q0[j]; e[j + 4] = (_Float16)q1[j]; }
        *((v8h*)sW + i) = e;
    }
    if (tid < HID_) sBiasK[tid] = (b_ih[tid] + b_hh[tid]) * TWOLOG2E;
    __syncthreads();

    _Float16* h = &sH[wave * RPW * HID_];

    // zero the wave-private h tile (8 KB = 512 x 16B)
    {
        v4f z = {0.f, 0.f, 0.f, 0.f};
        v4f* hz = (v4f*)h;
        for (int i = lane; i < (RPW * HID_) / 8; i += 32) hz[i] = z;
    }

    const int rowBase = blockIdx.x * RPB + wave * RPW;

#pragma unroll 1
    for (int t = 0; t < T_; ++t) {
        // ---- preload B-fragments: B[k][m] = hx[m][k]; per m-tile, 5 k-tiles ----
        // B as loaded here (contiguous k order, same as A): lane -> col m = lane&15, 16 consecutive k at hiHalf*16
        v16h bfrag[2][5];
#pragma unroll
        for (int mt = 0; mt < 2; ++mt) {
            const int m32 = mt * 16 + l15;
#pragma unroll
            for (int kt = 0; kt < 4; ++kt) {       // k-tiles 0..3 from h (LDS)
                const v8h* bp = (const v8h*)(h + m32 * HID_ + kt * 32 + hiHalf * 16);
                bfrag[mt][kt] = combine16(bp[0], bp[1]);
            }
            {   // k-tile 4: x_t straight from global (cols hiHalf*16..+15, 28..31 = 0)
                const float* xr = x + (size_t)(rowBase + m32) * (T_ * INPUT_)
                                  + t * INPUT_ + hiHalf * 16;
                const v4f* xv = (const v4f*)xr;    // 16B aligned (112 = 7*16)
                v4f q0 = xv[0], q1 = xv[1], q2 = xv[2];
                v4f q3 = {0.f, 0.f, 0.f, 0.f};
                if (hiHalf == 0) q3 = xv[3];
                v16h b;
#pragma unroll
                for (int j = 0; j < 4; ++j) {
                    b[j]      = (_Float16)q0[j];  b[j + 4]  = (_Float16)q1[j];
                    b[j + 8]  = (_Float16)q2[j];  b[j + 12] = (_Float16)q3[j];
                }
                bfrag[mt][4] = b;
                if (t + 1 < T_) __builtin_prefetch(xr + INPUT_, 0, 1);
            }
        }

        // ---- D'[n][m] = bias[n] + Wcat[n][:] . [h|x]^T[:, m], then tanh ----
#pragma unroll
        for (int nt = 0; nt < 8; ++nt) {
            v8f c0 = {}, c1 = {};                  // C starts at inline 0; bias folded below
#pragma unroll
            for (int kt = 0; kt < 5; ++kt) {
                // A as loaded here (contiguous k order, same as B): lane -> row n = nt*16+(lane&15),
                // 16 consecutive k at kt*32 + hiHalf*16
                const v8h* ap = (const v8h*)(sW + (nt * 16 + l15) * KCAT
                                             + kt * 32 + hiHalf * 16);
                v16h afrag = combine16(ap[0], ap[1]);
                c0 = __builtin_amdgcn_wmma_f32_16x16x32_f16(
                         false, afrag, false, bfrag[0][kt], (short)0, c0, false, false);
                c1 = __builtin_amdgcn_wmma_f32_16x16x32_f16(
                         false, afrag, false, bfrag[1][kt], (short)0, c1, false, false);
            }
            // D' layout: lane holds batch-row m = lane&15, n = nt*16 + hiHalf*8 + r
            const v4f* bkp = (const v4f*)(sBiasK + nt * 16 + hiHalf * 8);
            v4f bk0 = bkp[0], bk1 = bkp[1];
            v8h d0, d1;
#pragma unroll
            for (int r = 0; r < 8; ++r) {
                float bk = (r < 4) ? bk0[r & 3] : bk1[r & 3];
                // tanh(c + b) = 1 - 2/(exp2((c+b)*2log2e) + 1), bias pre-scaled
                float e0 = __builtin_amdgcn_exp2f(__builtin_fmaf(c0[r], TWOLOG2E, bk));
                float e1 = __builtin_amdgcn_exp2f(__builtin_fmaf(c1[r], TWOLOG2E, bk));
                d0[r] = (_Float16)(1.0f - 2.0f * __builtin_amdgcn_rcpf(e0 + 1.0f));
                d1[r] = (_Float16)(1.0f - 2.0f * __builtin_amdgcn_rcpf(e1 + 1.0f));
            }
            // one contiguous 16B store per m-tile
            *(v8h*)(h + (0 * 16 + l15) * HID_ + nt * 16 + hiHalf * 8) = d0;
            *(v8h*)(h + (1 * 16 + l15) * HID_ + nt * 16 + hiHalf * 8) = d1;
        }
    }

    // ---- final FC: out[b][o] = h . fc_w[o] + fc_b[o]; 32 rows x 10 outs per wave
#pragma unroll
    for (int mt = 0; mt < 2; ++mt) {
        const int m32 = mt * 16 + l15;
#pragma unroll
        for (int it = 0; it < 5; ++it) {
            int o = hiHalf + 2 * it;               // 0..9
            float acc = fc_b[o];
            const _Float16* hr = h + m32 * HID_;
            const float*    wr = fc_w + o * HID_;
#pragma unroll 4
            for (int k = 0; k < HID_; k += 8) {
                v8h hv = *(const v8h*)(hr + k);
                v4f w0 = *(const v4f*)(wr + k);
                v4f w1 = *(const v4f*)(wr + k + 4);
#pragma unroll
                for (int i = 0; i < 4; ++i) acc += (float)hv[i]     * w0[i];
#pragma unroll
                for (int i = 0; i < 4; ++i) acc += (float)hv[i + 4] * w1[i];
            }
            out[(size_t)(rowBase + m32) * OUT_ + o] = acc;
        }
    }
}

extern "C" void kernel_launch(void* const* d_in, const int* in_sizes, int n_in,
                              void* d_out, int out_size, void* d_ws, size_t ws_size,
                              hipStream_t stream) {
    const float* x    = (const float*)d_in[0];
    const float* w_ih = (const float*)d_in[1];
    const float* w_hh = (const float*)d_in[2];
    const float* b_ih = (const float*)d_in[3];
    const float* b_hh = (const float*)d_in[4];
    const float* fc_w = (const float*)d_in[5];
    const float* fc_b = (const float*)d_in[6];
    float* out = (float*)d_out;

    dim3 grid(BATCH_ / RPB);   // 512 workgroups
    dim3 block(128);           // 4 wave32s
    hipLaunchKernelGGL(rnn_fused_wmma, grid, block, 0, stream,
                       x, w_ih, w_hh, b_ih, b_hh, fc_w, fc_b, out);
}
